// MultiHeadAttention_45079976739218
// MI455X (gfx1250) — matrix operand order repaired and checked by reading, not compiled, not run on hardware
//
#include <hip/hip_runtime.h>
#include <hip/hip_bf16.h>

// ---------------------------------------------------------------------------
// MI455X (gfx1250) windowed multi-head attention.
// Compute-bound on the two GEMMs (~17 GFLOP) -> v_wmma_f32_16x16x32_f16 with
// f32 accumulation. Working set (~75 MB) is L2-resident (192 MB), so WMMA
// fragments are loaded straight from global (L2) using contiguous-k per-lane
// layouts (two 16B loads for A, one 32B load for B). K is hardcoded to 512 so
// all fragment addresses are immediate offsets off two base pointers (no
// per-iteration 64-bit address arithmetic, minimal VGPR pressure, no spills);
// #pragma unroll 4 lets the scheduler stagger s_wait_loadcnt across WMMAs.
// ---------------------------------------------------------------------------

typedef __attribute__((ext_vector_type(16))) _Float16 v16h;
typedef __attribute__((ext_vector_type(8)))  _Float16 v8h;
typedef __attribute__((ext_vector_type(8)))  float    v8f;

static constexpr int  NB   = 4;      // batch
static constexpr int  CIN  = 512;    // input channels
static constexpr int  WLEN = 2048;   // sequence length
static constexpr int  NH   = 8;      // heads
static constexpr int  AD   = 512;    // ATTN_DIM
static constexpr int  DH   = 64;     // head dim
static constexpr int  WIN  = 31;     // window width
static constexpr int  HWIN = 15;     // half window
static constexpr int  KDIM = 512;    // K for both GEMMs (CIN == AD == 512)
static constexpr float SCALE   = 1.0f / 64.0f;   // NUM_HEAD/ATTN_DIM
static constexpr float NEG_INF = -1.0e10f;

// ------------------------- fp32 -> fp16 prep kernels -----------------------

// input (n, c, 1, w) fp32 -> xh (n*w, c) fp16 row-major (tiled LDS transpose
// so both the global read and the global write are coalesced).
__global__ void mha_cvt_x_kernel(const float* __restrict__ in,
                                 _Float16* __restrict__ xh) {
  __shared__ float tile[32][33];
  const int n0 = blockIdx.z;
  const int i0 = blockIdx.x * 32;   // along w
  const int c0 = blockIdx.y * 32;   // along c
  const int li = threadIdx.x;       // 0..31
  const int lr = threadIdx.y;       // 0..7
#pragma unroll
  for (int k = 0; k < 4; ++k)
    tile[lr + 8 * k][li] =
        in[((size_t)n0 * CIN + (c0 + lr + 8 * k)) * WLEN + (i0 + li)];
  __syncthreads();
#pragma unroll
  for (int k = 0; k < 4; ++k)
    xh[((size_t)n0 * WLEN + (i0 + lr + 8 * k)) * CIN + (c0 + li)] =
        (_Float16)tile[li][lr + 8 * k];
}

__global__ void mha_cvt_w_kernel(const float* __restrict__ in,
                                 _Float16* __restrict__ out, int nElems) {
  int t = blockIdx.x * blockDim.x + threadIdx.x;
  if (t < nElems) out[t] = (_Float16)in[t];
}

#define MHA_WMMA(a_, b_, c_)                                                \
  __builtin_amdgcn_wmma_f32_16x16x32_f16(false, (a_), false, (b_),           \
                                         (short)0, (c_), false, false)

// ------------------------- WMMA GEMM mainloop ------------------------------
// One wave32 computes a 32(M) x 64(N) tile of A(MxK) @ B^T; B stored (N x K)
// row-major (the Linear weight layout); K == 512.
//
// Fragment layouts (contiguous k, same for A and B; V_WMMA_F32_16X16X32_F16):
//   A 16x32: lane holds row lane%16; halves 0..7 -> K = k0+16g+0..7,
//            halves 8..15 -> K = k0+16g+8..15           (two 16B loads)
//   B 32x16: lane holds col lane%16; K = k0+16g..+15     (one 32B load)
// With g folded into the base pointers, every load is baseptr + immediate.
__device__ __forceinline__ void mha_wmma_32x64(const _Float16* __restrict__ A,
                                               const _Float16* __restrict__ Bt,
                                               int rowBase, int colBase,
                                               v8f acc[2][4]) {
  const int lane = threadIdx.x & 31;
  const int lm   = lane & 15;
  const int g    = lane >> 4;
  const _Float16* aBase = A + (size_t)(rowBase + lm) * KDIM + 16 * g;
  const _Float16* bBase = Bt + (size_t)(colBase + lm) * KDIM + 16 * g;

#pragma unroll 4
  for (int k0 = 0; k0 < KDIM; k0 += 32) {
    union { v16h v; v8h h[2]; } a[2];
#pragma unroll
    for (int m = 0; m < 2; ++m) {
      a[m].h[0] = *reinterpret_cast<const v8h*>(aBase + m * 16 * KDIM + k0);
      a[m].h[1] = *reinterpret_cast<const v8h*>(aBase + m * 16 * KDIM + k0 + 8);
    }
    v16h b[4];
#pragma unroll
    for (int t = 0; t < 4; ++t)
      b[t] = *reinterpret_cast<const v16h*>(bBase + t * 16 * KDIM + k0);
#pragma unroll
    for (int m = 0; m < 2; ++m)
#pragma unroll
      for (int t = 0; t < 4; ++t)
        acc[m][t] = MHA_WMMA(a[m].v, b[t], acc[m][t]);
  }
}

// ------------------------- QKV projection GEMM -----------------------------
// temp(8192 x 1536) = xh(8192 x 512) @ W_in^T + b_in, scattered into
// q/k/v fp16 buffers laid out (n, h, token, d) for windowed attention.
// Block = 8 waves in a 4(M) x 2(N) grid -> 128 x 128 block tile.
__global__ void __launch_bounds__(256, 1)
mha_qkv_gemm_kernel(const _Float16* __restrict__ Xh,
                    const _Float16* __restrict__ Wh,
                    const float* __restrict__ bias,
                    _Float16* __restrict__ qh,
                    _Float16* __restrict__ kh,
                    _Float16* __restrict__ vh) {
  const int wave    = threadIdx.x >> 5;
  const int rowBase = blockIdx.x * 128 + (wave & 3) * 32;
  const int colBase = blockIdx.y * 128 + (wave >> 2) * 64;
  v8f acc[2][4] = {};
  mha_wmma_32x64(Xh, Wh, rowBase, colBase, acc);

  const int lane = threadIdx.x & 31;
  const int lm   = lane & 15;
  const int g    = lane >> 4;
#pragma unroll
  for (int m = 0; m < 2; ++m) {
#pragma unroll
    for (int t = 0; t < 4; ++t) {
      const int e  = colBase + t * 16 + lm;      // channel in [0, 1536)
      const int h  = e / 192;
      const int s  = (e % 192) / 64;             // 0=q 1=k 2=v
      const int dd = e % 64;
      _Float16* dst = (s == 0) ? qh : (s == 1) ? kh : vh;
      const float bv = bias[e];
#pragma unroll
      for (int r = 0; r < 8; ++r) {
        const int row = rowBase + m * 16 + 8 * g + r;  // token index
        const int n0  = row >> 11;                     // / 2048
        const int i   = row & (WLEN - 1);
        dst[(((size_t)n0 * NH + h) * WLEN + i) * DH + dd] =
            (_Float16)(acc[m][t][r] + bv);
      }
    }
  }
}

// ------------------------- banded softmax attention ------------------------
// One 64-thread block (2 waves) per (n, h, query i): thread t owns dim d = t.
__global__ void mha_attn_kernel(const _Float16* __restrict__ qh,
                                const _Float16* __restrict__ kh,
                                const _Float16* __restrict__ vh,
                                const float* __restrict__ mask,
                                _Float16* __restrict__ ao) {
  const int i  = blockIdx.x;
  const int h  = blockIdx.y;
  const int n0 = blockIdx.z;
  const int t  = threadIdx.x;  // 0..63
  __shared__ float sq[DH];
  __shared__ float sp[WIN];

  const size_t base = ((size_t)n0 * NH + h) * WLEN * DH;
  sq[t] = (float)qh[base + (size_t)i * DH + t];
  __syncthreads();

  if (t < WIN) {
    const int j = i - HWIN + t;
    float s = NEG_INF;
    if (j >= 0 && j < WLEN && mask[(size_t)n0 * WLEN + j] > 0.5f) {
      const _Float16* kp = kh + base + (size_t)j * DH;
      float d0 = 0.f;
#pragma unroll
      for (int d = 0; d < DH; ++d) d0 += sq[d] * (float)kp[d];
      s = d0 * SCALE;
    }
    sp[t] = s;
  }
  __syncthreads();

  if (t == 0) {
    float mx = NEG_INF;
#pragma unroll
    for (int u = 0; u < WIN; ++u) mx = fmaxf(mx, sp[u]);
    float e[WIN];
    float sum = 0.f;
#pragma unroll
    for (int u = 0; u < WIN; ++u) {
      float v0 = (sp[u] <= 0.5f * NEG_INF) ? 0.f : __expf(sp[u] - mx);
      e[u] = v0;
      sum += v0;
    }
    const float inv = (sum > 0.f) ? (1.f / sum) : 0.f;
#pragma unroll
    for (int u = 0; u < WIN; ++u) sp[u] = e[u] * inv;
  }
  __syncthreads();

  float acc = 0.f;
#pragma unroll
  for (int u = 0; u < WIN; ++u) {
    const int j = i - HWIN + u;
    if (j >= 0 && j < WLEN)
      acc += sp[u] * (float)vh[base + (size_t)j * DH + t];
  }
  // Row-major (token, channel) -> directly the A matrix of the FC GEMM.
  ao[((size_t)n0 * WLEN + i) * AD + (h * DH + t)] = (_Float16)acc;
}

// ------------------------- output FC GEMM ----------------------------------
// out2(8192 x 512) = ao @ W_fc^T + b_fc, written as (n, 512, 1, w) fp32.
__global__ void __launch_bounds__(256, 1)
mha_fc_gemm_kernel(const _Float16* __restrict__ Ah,
                   const _Float16* __restrict__ Wh,
                   const float* __restrict__ bias,
                   float* __restrict__ out) {
  const int wave    = threadIdx.x >> 5;
  const int rowBase = blockIdx.x * 128 + (wave & 3) * 32;
  const int colBase = blockIdx.y * 128 + (wave >> 2) * 64;
  v8f acc[2][4] = {};
  mha_wmma_32x64(Ah, Wh, rowBase, colBase, acc);

  const int lane = threadIdx.x & 31;
  const int lm   = lane & 15;
  const int g    = lane >> 4;
#pragma unroll
  for (int m = 0; m < 2; ++m) {
#pragma unroll
    for (int t = 0; t < 4; ++t) {
      const int od = colBase + t * 16 + lm;
      const float bv = bias[od];
#pragma unroll
      for (int r = 0; r < 8; ++r) {
        const int row = rowBase + m * 16 + 8 * g + r;  // token
        const int n0  = row >> 11;
        const int i   = row & (WLEN - 1);
        out[((size_t)n0 * AD + od) * WLEN + i] = acc[m][t][r] + bv;
      }
    }
  }
}

// ------------------------- launch ------------------------------------------
extern "C" void kernel_launch(void* const* d_in, const int* in_sizes, int n_in,
                              void* d_out, int out_size, void* d_ws, size_t ws_size,
                              hipStream_t stream) {
  (void)in_sizes; (void)n_in; (void)out_size; (void)ws_size;
  const float* input = (const float*)d_in[0];
  const float* mask  = (const float*)d_in[1];
  const float* W_in  = (const float*)d_in[2];
  const float* b_in  = (const float*)d_in[3];
  const float* W_fc  = (const float*)d_in[4];
  const float* b_fc  = (const float*)d_in[5];
  float* out = (float*)d_out;

  // Workspace carve-out (~42 MB total, 256B aligned slices).
  char* ws = (char*)d_ws;
  size_t off = 0;
  auto carve = [&](size_t bytes) -> void* {
    void* p = ws + off;
    off += (bytes + 255) & ~(size_t)255;
    return p;
  };
  _Float16* xh   = (_Float16*)carve((size_t)NB * WLEN * CIN * 2);      // 8 MB
  _Float16* winh = (_Float16*)carve((size_t)3 * AD * CIN * 2);         // 1.5 MB
  _Float16* wfch = (_Float16*)carve((size_t)AD * AD * 2);              // 0.5 MB
  _Float16* qh   = (_Float16*)carve((size_t)NB * NH * WLEN * DH * 2);  // 8 MB
  _Float16* kh   = (_Float16*)carve((size_t)NB * NH * WLEN * DH * 2);  // 8 MB
  _Float16* vh   = (_Float16*)carve((size_t)NB * NH * WLEN * DH * 2);  // 8 MB
  _Float16* ao   = (_Float16*)carve((size_t)NB * WLEN * AD * 2);       // 8 MB

  // 1) fp32 -> fp16 staging
  mha_cvt_x_kernel<<<dim3(WLEN / 32, CIN / 32, NB), dim3(32, 8), 0, stream>>>(input, xh);
  mha_cvt_w_kernel<<<(3 * AD * CIN + 255) / 256, 256, 0, stream>>>(W_in, winh, 3 * AD * CIN);
  mha_cvt_w_kernel<<<(AD * AD + 255) / 256, 256, 0, stream>>>(W_fc, wfch, AD * AD);

  // 2) QKV projection GEMM: M=8192, N=1536, K=512; 128x128 block tile
  mha_qkv_gemm_kernel<<<dim3((NB * WLEN) / 128, (3 * AD) / 128), 256, 0, stream>>>(
      xh, winh, b_in, qh, kh, vh);

  // 3) banded softmax attention: one block per (query, head, batch)
  mha_attn_kernel<<<dim3(WLEN, NH, NB), DH, 0, stream>>>(qh, kh, vh, mask, ao);

  // 4) output FC GEMM: M=8192, N=512, K=512; 128x128 block tile
  mha_fc_gemm_kernel<<<dim3((NB * WLEN) / 128, AD / 128), 256, 0, stream>>>(
      ao, wfch, b_fc, out);
}
